// EGATLayer_90486370992246
// MI455X (gfx1250) — matrix operand order repaired and checked by reading, not compiled, not run on hardware
//
#include <hip/hip_runtime.h>
#include <hip/hip_bf16.h>

// ---------------------------------------------------------------------------
// EGAT layer for MI455X (gfx1250, wave32).
// Heavy GEMMs (E_star = ME@E, se2 = (MH*g)@MH^T) via v_wmma_f32_16x16x32_bf16
// with exact 0/1 masks in bf16 and hi/lo bf16 splitting of the float operand
// (f32 accumulate) for near-fp32 accuracy.
// ---------------------------------------------------------------------------

typedef __attribute__((ext_vector_type(16))) __bf16 v16bf;
typedef __attribute__((ext_vector_type(8)))  float  v8f;

#define Bb 2
#define Nn 256
#define Mm 1024
#define Dd 64
#define NEGV (-9.0e15f)

// ---- helpers --------------------------------------------------------------

__device__ __forceinline__ v8f zero8() {
  v8f z;
#pragma unroll
  for (int i = 0; i < 8; ++i) z[i] = 0.0f;
  return z;
}

__device__ __forceinline__ void put4(v16bf& v, int o, float4 u) {
  v[o + 0] = (__bf16)u.x; v[o + 1] = (__bf16)u.y;
  v[o + 2] = (__bf16)u.z; v[o + 3] = (__bf16)u.w;
}

__device__ __forceinline__ float reduce64(float v, float* red) {
  int t = threadIdx.x;
  red[t] = v; __syncthreads();
#pragma unroll
  for (int s = 32; s > 0; s >>= 1) {
    if (t < s) red[t] += red[t + s];
    __syncthreads();
  }
  float r = red[0]; __syncthreads();
  return r;
}

__device__ __forceinline__ float redMax256(float v, float* red) {
  int t = threadIdx.x;
  red[t] = v; __syncthreads();
#pragma unroll
  for (int s = 128; s > 0; s >>= 1) {
    if (t < s) red[t] = fmaxf(red[t], red[t + s]);
    __syncthreads();
  }
  float r = red[0]; __syncthreads();
  return r;
}

__device__ __forceinline__ float redSum256(float v, float* red) {
  int t = threadIdx.x;
  red[t] = v; __syncthreads();
#pragma unroll
  for (int s = 128; s > 0; s >>= 1) {
    if (t < s) red[t] += red[t + s];
    __syncthreads();
  }
  float r = red[0]; __syncthreads();
  return r;
}

// ---- k_prep: H_trans, E_trans, hA1, hA2, eB1, eB2, g ----------------------
// grid = B*(N+M) blocks of 64 threads; one row each.

__global__ __launch_bounds__(64) void k_prep(
    const float* __restrict__ H, const float* __restrict__ E,
    const float* __restrict__ W_H, const float* __restrict__ b_H,
    const float* __restrict__ W_E, const float* __restrict__ b_E,
    const float* __restrict__ a, const float* __restrict__ b_attn,
    float* __restrict__ Ht, float* __restrict__ Et,
    float* __restrict__ hA1, float* __restrict__ hA2,
    float* __restrict__ eB1, float* __restrict__ eB2, float* __restrict__ g) {
  __shared__ float xin[64];
  __shared__ float red[64];
  int r = blockIdx.x;
  int f = threadIdx.x;
  if (r < Bb * Nn) {                       // H rows
    xin[f] = H[(size_t)r * Dd + f];
    __syncthreads();
    float acc = b_H[f];
#pragma unroll 8
    for (int d = 0; d < Dd; ++d) acc += xin[d] * W_H[f * Dd + d];
    Ht[(size_t)r * Dd + f] = acc;
    float s1 = reduce64(acc * a[f], red);          // a1
    float s2 = reduce64(acc * a[Dd + f], red);     // a2
    float s3 = reduce64(xin[f] * b_attn[2 * Dd + f], red);  // g = H @ b3
    if (f == 0) { hA1[r] = s1; hA2[r] = s2; g[r] = s3; }
  } else {                                 // E rows
    int r2 = r - Bb * Nn;
    xin[f] = E[(size_t)r2 * Dd + f];
    __syncthreads();
    float acc = b_E[f];
#pragma unroll 8
    for (int d = 0; d < Dd; ++d) acc += xin[d] * W_E[f * Dd + d];
    Et[(size_t)r2 * Dd + f] = acc;
    float s1 = reduce64(acc * b_attn[f], red);       // b1
    float s2 = reduce64(acc * b_attn[Dd + f], red);  // b2
    if (f == 0) { eB1[r2] = s1; eB2[r2] = s2; }
  }
}

// ---- k_estar: E_star[b,k,f] = sum_m ME[b,k,m]*E[b,m,f]  (WMMA bf16) -------
// grid = (512, B), 256 threads (8 waves x 16-row slabs = 128 rows/block).
// K is tiled in chunks of 256; E chunk staged transposed in LDS as bf16 hi/lo.

#define KC   256
#define LSTR (KC + 16)   // padded LDS row stride (elements)

__global__ __launch_bounds__(256) void k_estar(
    const float* __restrict__ ME, const float* __restrict__ E,
    float* __restrict__ E_star) {
  extern __shared__ v16bf smem_v[];
  __bf16* Eh = (__bf16*)smem_v;           // [64][LSTR] hi part
  __bf16* El = Eh + Dd * LSTR;            // [64][LSTR] lo part

  const int b    = blockIdx.y;
  const int tid  = threadIdx.x;
  const int wave = tid >> 5;
  const int lane = tid & 31;
  const int l15  = lane & 15;
  const int hi16 = lane >> 4;             // 0: lanes 0-15, 1: lanes 16-31

  const size_t NK = (size_t)Nn * Nn;      // 65536 rows per batch
  const size_t rowbase = (size_t)blockIdx.x * 128 + wave * 16;
  const float* MEb = ME + (size_t)b * NK * Mm;
  const float* Eb  = E  + (size_t)b * Mm * Dd;
  const float* Arow = MEb + (rowbase + l15) * Mm;

  v8f acc[4];
#pragma unroll
  for (int t = 0; t < 4; ++t) acc[t] = zero8();

  for (int kc = 0; kc < Mm / KC; ++kc) {
    // stage E chunk (KC x 64) transposed into LDS, split hi/lo bf16
    {
      int f  = tid & 63;
      int mg = tid >> 6;                  // 0..3
#pragma unroll 4
      for (int mi = 0; mi < 64; ++mi) {
        int ml = mg * 64 + mi;
        float x = Eb[(size_t)(kc * KC + ml) * Dd + f];
        __bf16 h = (__bf16)x;
        float  r = x - (float)h;
        Eh[f * LSTR + ml] = h;
        El[f * LSTR + ml] = (__bf16)r;
      }
    }
    __syncthreads();

#pragma unroll 2
    for (int ks = 0; ks < KC / 32; ++ks) {
      int kb = kc * KC + ks * 32;
      // A operand: 16x32 of ME (0/1 -> exact bf16).
      // lane<16 holds K {0..15}, lane>=16 holds K {16..31}.
      int seg0 = kb + (hi16 ? 16 : 0);
      float4 u0 = *(const float4*)(Arow + seg0);
      float4 u1 = *(const float4*)(Arow + seg0 + 4);
      float4 u2 = *(const float4*)(Arow + seg0 + 8);
      float4 u3 = *(const float4*)(Arow + seg0 + 12);
      v16bf av;
      put4(av, 0, u0); put4(av, 4, u1); put4(av, 8, u2); put4(av, 12, u3);
      __builtin_prefetch(Arow + kb + 64, 0, 1);   // global_prefetch next A

      int kl = ks * 32 + (hi16 ? 16 : 0);         // B: lanes<16 K0-15, >=16 K16-31
#pragma unroll
      for (int ft = 0; ft < 4; ++ft) {
        int f = ft * 16 + l15;
        v16bf bh = *(const v16bf*)(Eh + f * LSTR + kl);
        v16bf bl = *(const v16bf*)(El + f * LSTR + kl);
        acc[ft] = __builtin_amdgcn_wmma_f32_16x16x32_bf16(
            false, av, false, bh, (short)0, acc[ft], false, false);
        acc[ft] = __builtin_amdgcn_wmma_f32_16x16x32_bf16(
            false, av, false, bl, (short)0, acc[ft], false, false);
      }
    }
    __syncthreads();
  }

  // C/D layout: VGPR vr, lane l -> row vr + 8*(l>=16), col l&15
#pragma unroll
  for (int ft = 0; ft < 4; ++ft) {
    int f = ft * 16 + l15;
#pragma unroll
    for (int vr = 0; vr < 8; ++vr) {
      size_t row = rowbase + vr + 8 * hi16;
      E_star[((size_t)b * NK + row) * Dd + f] = acc[ft][vr];
    }
  }
}

// ---- k_e3: e3[row] = dot(E_star[row,:], a3) ------------------------------

__global__ __launch_bounds__(256) void k_e3(
    const float* __restrict__ E_star, const float* __restrict__ a,
    float* __restrict__ e3) {
  __shared__ float a3s[Dd];
  if (threadIdx.x < Dd) a3s[threadIdx.x] = a[2 * Dd + threadIdx.x];
  __syncthreads();
  size_t row = (size_t)blockIdx.x * 256 + threadIdx.x;   // B*N*N rows
  const float4* p = (const float4*)(E_star + row * Dd);
  float s = 0.0f;
#pragma unroll
  for (int i = 0; i < Dd / 4; ++i) {
    float4 u = p[i];
    s += u.x * a3s[4 * i] + u.y * a3s[4 * i + 1] +
         u.z * a3s[4 * i + 2] + u.w * a3s[4 * i + 3];
  }
  e3[row] = s;
}

// ---- k_se: se2[b,p,q] = sum_n MH[b,p,n]*g[b,n]*MH[b,q,n]  (WMMA bf16) -----
// grid = (8, B), 256 threads. A = MH*g (hi/lo split) held in registers;
// B tiles = contiguous MH rows (exact 0/1 bf16).

__global__ __launch_bounds__(256) void k_se(
    const float* __restrict__ MH, const float* __restrict__ g,
    float* __restrict__ se2) {
  const int b    = blockIdx.y;
  const int tid  = threadIdx.x;
  const int wave = tid >> 5;
  const int lane = tid & 31;
  const int l15  = lane & 15;
  const int hi16 = lane >> 4;
  const int prow = blockIdx.x * 128 + wave * 16 + l15;
  const float* MHb = MH + (size_t)b * Mm * Nn;
  const float* gb  = g + b * Nn;
  const float* pr  = MHb + (size_t)prow * Nn;

  // Pre-scaled A in registers: 8 k-steps x v16bf (hi and lo).
  v16bf Ah[8], Al[8];
#pragma unroll
  for (int ks = 0; ks < 8; ++ks) {
    int seg0 = ks * 32 + (hi16 ? 16 : 0);
#pragma unroll
    for (int s = 0; s < 2; ++s) {
      int base = seg0 + s * 8;
#pragma unroll
      for (int j = 0; j < 8; ++j) {
        float m  = pr[base + j];
        float gv = gb[base + j];
        __bf16 gh = (__bf16)gv;
        __bf16 gl = (__bf16)(gv - (float)gh);
        bool on = (m != 0.0f);
        Ah[ks][s * 8 + j] = on ? gh : (__bf16)0.0f;
        Al[ks][s * 8 + j] = on ? gl : (__bf16)0.0f;
      }
    }
  }

  for (int qt = 0; qt < Mm / 16; ++qt) {
    v8f acc = zero8();
    int qcol = qt * 16 + l15;
    const float* Brow = MHb + (size_t)qcol * Nn;
#pragma unroll 2
    for (int ks = 0; ks < 8; ++ks) {
      int kb = ks * 32 + (hi16 ? 16 : 0);
      float4 w0 = *(const float4*)(Brow + kb);
      float4 w1 = *(const float4*)(Brow + kb + 4);
      float4 w2 = *(const float4*)(Brow + kb + 8);
      float4 w3 = *(const float4*)(Brow + kb + 12);
      v16bf bv;
      put4(bv, 0, w0); put4(bv, 4, w1); put4(bv, 8, w2); put4(bv, 12, w3);
      acc = __builtin_amdgcn_wmma_f32_16x16x32_bf16(
          false, Ah[ks], false, bv, (short)0, acc, false, false);
      acc = __builtin_amdgcn_wmma_f32_16x16x32_bf16(
          false, Al[ks], false, bv, (short)0, acc, false, false);
    }
#pragma unroll
    for (int vr = 0; vr < 8; ++vr) {
      int row = blockIdx.x * 128 + wave * 16 + vr + 8 * hi16;
      se2[((size_t)b * Mm + row) * Mm + qcol] = acc[vr];
    }
  }
}

// ---- k_H: per (b,i) row softmax + H_prime + Hm ---------------------------

__global__ __launch_bounds__(256) void k_H(
    const float* __restrict__ AH, const float* __restrict__ e3,
    const float* __restrict__ hA1, const float* __restrict__ hA2,
    const float* __restrict__ Ht, const float* __restrict__ E_star,
    float* __restrict__ Hp, float* __restrict__ Hm) {
  __shared__ float red[256];
  __shared__ float wsh[256];
  __shared__ float pp[256];
  __shared__ float pm[256];
  int i = blockIdx.x, b = blockIdx.y, j = threadIdx.x;

  float s = hA1[b * Nn + i] + hA2[b * Nn + j] +
            e3[(size_t)b * Nn * Nn + (size_t)i * Nn + j];
  s = (s > 0.0f) ? s : 0.2f * s;                 // leaky_relu 0.2
  bool mk = AH[(size_t)i * Nn + j] > 0.0f;       // mask from batch 0
  s = mk ? s : NEGV;
  float mx  = redMax256(s, red);
  float e   = __expf(s - mx);
  float sum = redSum256(e, red);
  wsh[j] = mk ? (e / sum) : 0.0f;
  __syncthreads();

  int f = j & 63, c = j >> 6;
  const float* Htb = Ht + (size_t)b * Nn * Dd;
  const float* Esb = E_star + ((size_t)b * Nn * Nn + (size_t)i * Nn) * Dd;
  float ap = 0.0f, am = 0.0f;
#pragma unroll 4
  for (int jj = 0; jj < 64; ++jj) {
    int jq = c * 64 + jj;
    float w  = wsh[jq];
    float h  = Htb[(size_t)jq * Dd + f];
    float es = Esb[(size_t)jq * Dd + f];
    ap += w * h;
    am += w * h * es;
  }
  pp[j] = ap; pm[j] = am;
  __syncthreads();
  if (c == 0) {
    float sp = pp[f] + pp[64 + f] + pp[128 + f] + pp[192 + f];
    float sm = pm[f] + pm[64 + f] + pm[128 + f] + pm[192 + f];
    Hp[((size_t)b * Nn + i) * Dd + f] = sp;
    Hm[((size_t)b * Nn + i) * Dd + f] = sm;
  }
}

// ---- k_E: per (b,p) row softmax + E_prime --------------------------------

__global__ __launch_bounds__(256) void k_E(
    const float* __restrict__ AE, const float* __restrict__ se2,
    const float* __restrict__ eB1, const float* __restrict__ eB2,
    const float* __restrict__ Et, float* __restrict__ Ep) {
  __shared__ float red[256];
  __shared__ float sq[Mm];     // scores, then unnormalized masked weights
  __shared__ float part[256];
  int p = blockIdx.x, b = blockIdx.y, t = threadIdx.x;

  float eb1 = eB1[b * Mm + p];
  float lmax = NEGV;
#pragma unroll
  for (int c = 0; c < 4; ++c) {
    int q = c * 256 + t;
    float s = eb1 + eB2[b * Mm + q] +
              se2[((size_t)b * Mm + p) * Mm + q];
    s = (s > 0.0f) ? s : 0.2f * s;
    bool mk = AE[(size_t)p * Mm + q] > 0.0f;     // mask from batch 0
    s = mk ? s : NEGV;
    sq[q] = s;
    lmax = fmaxf(lmax, s);
  }
  float mx = redMax256(lmax, red);
  float lsum = 0.0f;
#pragma unroll
  for (int c = 0; c < 4; ++c) {
    int q = c * 256 + t;
    float e = __expf(sq[q] - mx);
    bool mk = AE[(size_t)p * Mm + q] > 0.0f;
    lsum += e;                                   // sum over ALL exps
    sq[q] = mk ? e : 0.0f;                       // masked numerator
  }
  float sum = redSum256(lsum, red);

  int f = t & 63, c4 = t >> 6;
  float aacc = 0.0f;
#pragma unroll 4
  for (int jj = 0; jj < 256; ++jj) {
    int q = c4 * 256 + jj;
    aacc += sq[q] * Et[((size_t)b * Mm + q) * Dd + f];
  }
  part[t] = aacc;
  __syncthreads();
  if (c4 == 0) {
    float v = part[f] + part[64 + f] + part[128 + f] + part[192 + f];
    Ep[((size_t)b * Mm + p) * Dd + f] = v / sum;
  }
}

// ---- launch ---------------------------------------------------------------

extern "C" void kernel_launch(void* const* d_in, const int* in_sizes, int n_in,
                              void* d_out, int out_size, void* d_ws, size_t ws_size,
                              hipStream_t stream) {
  const float* H      = (const float*)d_in[0];
  const float* E      = (const float*)d_in[1];
  const float* AH     = (const float*)d_in[2];
  const float* AE     = (const float*)d_in[3];
  const float* ME     = (const float*)d_in[4];
  const float* MH     = (const float*)d_in[5];
  const float* W_H    = (const float*)d_in[6];
  const float* b_H    = (const float*)d_in[7];
  const float* W_E    = (const float*)d_in[8];
  const float* b_E    = (const float*)d_in[9];
  const float* a      = (const float*)d_in[10];
  const float* b_attn = (const float*)d_in[11];

  // Workspace layout (bytes); total ~43.1 MB.
  char* ws = (char*)d_ws;
  float* Ht    = (float*)(ws + 0);          //  131072 B
  float* Et    = (float*)(ws + 131072);     //  524288 B
  float* hA1   = (float*)(ws + 655360);     //    2048 B
  float* hA2   = (float*)(ws + 657408);     //    2048 B
  float* eB1   = (float*)(ws + 659456);     //    8192 B
  float* eB2   = (float*)(ws + 667648);     //    8192 B
  float* g     = (float*)(ws + 675840);     //    2048 B
  float* e3    = (float*)(ws + 677888);     //  524288 B
  float* se2   = (float*)(ws + 1202176);    // 8388608 B
  float* Estar = (float*)(ws + 9590784);    // 33554432 B -> 43145216 total

  float* out = (float*)d_out;
  float* Hp = out;                          // (B,N,D)  32768
  float* Ep = out + Bb * Nn * Dd;           // (B,M,D) 131072
  float* Hm = out + Bb * Nn * Dd + Bb * Mm * Dd;  // (B,N,D)

  k_prep<<<Bb * (Nn + Mm), 64, 0, stream>>>(H, E, W_H, b_H, W_E, b_E, a, b_attn,
                                            Ht, Et, hA1, hA2, eB1, eB2, g);

  size_t lds_bytes = (size_t)2 * Dd * LSTR * sizeof(__bf16);  // 69632
  k_estar<<<dim3(512, Bb), 256, lds_bytes, stream>>>(ME, E, Estar);

  k_e3<<<(Bb * Nn * Nn) / 256, 256, 0, stream>>>(Estar, a, e3);

  k_se<<<dim3(Mm / 128, Bb), 256, 0, stream>>>(MH, g, se2);

  k_H<<<dim3(Nn, Bb), 256, 0, stream>>>(AH, e3, hA1, hA2, Ht, Estar, Hp, Hm);

  k_E<<<dim3(Mm, Bb), 256, 0, stream>>>(AE, se2, eB1, eB2, Et, Ep);

  (void)in_sizes; (void)n_in; (void)out_size; (void)ws_size;
}
